// GINNet_58205396795403
// MI455X (gfx1250) — matrix operand order repaired and checked by reading, not compiled, not run on hardware
//
#include <hip/hip_runtime.h>

#define N_NODES   100000
#define N_EDGES   3200000
#define D         64
#define N_LAYERS  4
#define OUT_STRIDE (D + N_LAYERS * D)   // 320
#define MT        (N_NODES / 16)        // 6250 row tiles (exact)

typedef __attribute__((ext_vector_type(16))) __bf16 v16bf;
typedef __attribute__((ext_vector_type(8)))  __bf16 v8bf;
typedef __attribute__((ext_vector_type(8)))  float  v8f;

// float -> bf16, round-to-nearest-even
static __device__ __forceinline__ __bf16 f2bf(float f) {
    union { float f; unsigned u; } a;
    a.f = f;
    unsigned u = a.u;
    u += 0x7FFFu + ((u >> 16) & 1u);
    union { unsigned short s; __bf16 b; } r;
    r.s = (unsigned short)(u >> 16);
    return r.b;
}

// ---------------------------------------------------------------------------
// Copy x into columns [0,64) of the [N,320] output.
__global__ void copy_x_kernel(const float* __restrict__ x, float* __restrict__ out) {
    int i = blockIdx.x * blockDim.x + threadIdx.x;
    if (i >= N_NODES * D) return;
    int r = i >> 6, c = i & 63;
    out[(long)r * OUT_STRIDE + c] = x[i];
}

// H = (1 + eps) * feat      (feat has row stride `stride`)
__global__ void init_h_kernel(const float* __restrict__ feat, int stride,
                              const float* __restrict__ eps, float* __restrict__ H) {
    int i = blockIdx.x * blockDim.x + threadIdx.x;
    if (i >= N_NODES * D) return;
    int r = i >> 6, c = i & 63;
    H[i] = (1.0f + eps[0]) * feat[(long)r * stride + c];
}

// One wave per edge: H[dst] += feat[src]  (64 floats, native f32 atomics)
__global__ void scatter_kernel(const float* __restrict__ feat, int stride,
                               const int* __restrict__ src, const int* __restrict__ dst,
                               float* __restrict__ H) {
    int warp = (blockIdx.x * blockDim.x + threadIdx.x) >> 5;
    int lane = threadIdx.x & 31;
    if (warp >= N_EDGES) return;
    int s = src[warp];
    int d = dst[warp];
    const float* fs = feat + (long)s * stride;
    float*       hd = H    + (long)d * D;
    float v0 = fs[lane];
    float v1 = fs[lane + 32];
    __hip_atomic_fetch_add(hd + lane,      v0, __ATOMIC_RELAXED, __HIP_MEMORY_SCOPE_AGENT);
    __hip_atomic_fetch_add(hd + lane + 32, v1, __ATOMIC_RELAXED, __HIP_MEMORY_SCOPE_AGENT);
}

__global__ void zero_stats_kernel(float* __restrict__ stats) {
    if (threadIdx.x < 128) stats[threadIdx.x] = 0.0f;
}

// Plain f32 -> bf16 convert (stage-1 GEMM input, no BN)
__global__ void cvt_bf16_kernel(const float* __restrict__ src, __bf16* __restrict__ dst) {
    int i = blockIdx.x * blockDim.x + threadIdx.x;
    if (i >= N_NODES * D) return;
    dst[i] = f2bf(src[i]);
}

// dst = bf16(relu(src*scale[c] + shift[c]))  — BN+ReLU materialized for next GEMM
__global__ void bnapply_bf16_kernel(const float* __restrict__ src,
                                    const float* __restrict__ scale,
                                    const float* __restrict__ shift,
                                    __bf16* __restrict__ dst) {
    int i = blockIdx.x * blockDim.x + threadIdx.x;
    if (i >= N_NODES * D) return;
    int c = i & 63;
    float v = src[i] * scale[c] + shift[c];
    v = v > 0.0f ? v : 0.0f;
    dst[i] = f2bf(v);
}

// Pre-pack one 64x64 f32 weight matrix into WMMA B-fragment order (bf16):
// frag[((nt*2 + ks)*32 + lane)*16 + e] = bf16( W[(ks*32 + 8*lane4 + (e&7) + 16*(e>>3))*64 + nt*16 + (lane&15)] )
__global__ void prep_w_kernel(const float* __restrict__ W, __bf16* __restrict__ frag) {
    int idx = blockIdx.x * blockDim.x + threadIdx.x;   // 0..4095
    if (idx >= 4096) return;
    int e    = idx & 15;
    int lane = (idx >> 4) & 31;
    int ks   = (idx >> 9) & 1;
    int nt   = idx >> 10;
    int kB   = ks * 32 + ((lane & 16) ? 8 : 0) + (e & 7) + ((e & 8) ? 16 : 0);
    int col  = nt * 16 + (lane & 15);
    frag[idx] = f2bf(W[kB * 64 + col]);
}

// out[M,64] = Abf[M,64] @ W + bias (bf16 WMMA, f32 accum).
// One wave per 16x64 output strip: A loaded once (4x b128), 8 WMMAs.
// Epilogue accumulates per-column sum / sum^2 for the following BatchNorm.
__global__ void __launch_bounds__(256)
gemm64_kernel(const __bf16* __restrict__ Abf,
              const __bf16* __restrict__ Wfrag,   // [4][2][32][16] bf16
              const float* __restrict__ bias,
              float* __restrict__ out,
              float* __restrict__ colsum, float* __restrict__ colsq) {
    int wv   = threadIdx.x >> 5;
    int lane = threadIdx.x & 31;
    int mt   = blockIdx.x * 8 + wv;
    if (mt >= MT) return;                       // uniform per wave

    int rowA = mt * 16 + (lane & 15);
    int kbA  = (lane & 16) ? 8 : 0;             // ISA 16-bit A layout K-group
    const __bf16* arow = Abf + (long)rowA * 64;

    // A fragments for both 32-wide K-steps: two contiguous 8xbf16 runs each.
    v16bf a[2];
    #pragma unroll
    for (int ks = 0; ks < 2; ++ks) {
        v8bf lo = *(const v8bf*)(arow + ks * 32 + kbA);
        v8bf hi = *(const v8bf*)(arow + ks * 32 + 16 + kbA);
        a[ks] = __builtin_shufflevector(lo, hi,
                0, 1, 2, 3, 4, 5, 6, 7, 8, 9, 10, 11, 12, 13, 14, 15);
    }

    int rbase = mt * 16 + ((lane >> 4) << 3);   // C layout: vgpr r -> row r (+8 hi half)
    #pragma unroll
    for (int nt = 0; nt < 4; ++nt) {
        v16bf b0 = *(const v16bf*)(Wfrag + ((nt * 2 + 0) * 32 + lane) * 16);
        v16bf b1 = *(const v16bf*)(Wfrag + ((nt * 2 + 1) * 32 + lane) * 16);
        v8f c = {};
        c = __builtin_amdgcn_wmma_f32_16x16x32_bf16(false, a[0], false, b0,
                                                    (short)0, c, false, false);
        c = __builtin_amdgcn_wmma_f32_16x16x32_bf16(false, a[1], false, b1,
                                                    (short)0, c, false, false);

        int ncol = nt * 16 + (lane & 15);
        float bi = bias[ncol];
        float s1 = 0.0f, s2 = 0.0f;
        #pragma unroll
        for (int r = 0; r < 8; ++r) {
            float v = c[r] + bi;
            out[(long)(rbase + r) * 64 + ncol] = v;
            s1 += v;
            s2 += v * v;
        }
        __hip_atomic_fetch_add(colsum + ncol, s1, __ATOMIC_RELAXED, __HIP_MEMORY_SCOPE_AGENT);
        __hip_atomic_fetch_add(colsq  + ncol, s2, __ATOMIC_RELAXED, __HIP_MEMORY_SCOPE_AGENT);
    }
}

// Per-channel BN folding: scale = g*rsqrt(var+eps), shift = beta - mean*scale
__global__ void bnfin_kernel(const float* __restrict__ colsum, const float* __restrict__ colsq,
                             const float* __restrict__ g, const float* __restrict__ be,
                             float* __restrict__ scale, float* __restrict__ shift) {
    int ch = threadIdx.x;
    if (ch >= D) return;
    const float invN = 1.0f / (float)N_NODES;
    float mean = colsum[ch] * invN;
    float var  = colsq[ch] * invN - mean * mean;
    float sc   = g[ch] * rsqrtf(var + 1e-5f);
    scale[ch]  = sc;
    shift[ch]  = be[ch] - mean * sc;
}

// o = relu(bn(T)) written into output slab columns [colbase, colbase+64)
__global__ void apply_out_kernel(const float* __restrict__ T,
                                 const float* __restrict__ scale, const float* __restrict__ shift,
                                 float* __restrict__ out, int colbase) {
    int i = blockIdx.x * blockDim.x + threadIdx.x;
    if (i >= N_NODES * D) return;
    int r = i >> 6, c = i & 63;
    float v = T[i] * scale[c] + shift[c];
    v = v > 0.0f ? v : 0.0f;
    out[(long)r * OUT_STRIDE + colbase + c] = v;
}

// ---------------------------------------------------------------------------
extern "C" void kernel_launch(void* const* d_in, const int* in_sizes, int n_in,
                              void* d_out, int out_size, void* d_ws, size_t ws_size,
                              hipStream_t stream) {
    const float* x   = (const float*)d_in[0];
    const int*   ei  = (const int*)d_in[1];
    const int*   src = ei;              // edge_index[0]
    const int*   dst = ei + N_EDGES;    // edge_index[1]
    float* out = (float*)d_out;

    float* ws     = (float*)d_ws;
    float* H      = ws;                                // [N,64] f32
    float* T      = ws + (size_t)N_NODES * D;          // [N,64] f32
    float* stats  = ws + (size_t)2 * N_NODES * D;      // 256 f32
    float* colsum = stats;
    float* colsq  = stats + 64;
    float* scale  = stats + 128;
    float* shift  = stats + 192;
    __bf16* Abf   = (__bf16*)(stats + 256);            // [N,64] bf16 (16B-aligned)
    __bf16* Wfrag = Abf + (size_t)N_NODES * D;         // 12 x 4096 bf16 fragments

    const int elemBlocks    = (N_NODES * D + 255) / 256;   // 25000
    const int gemmBlocks    = (MT + 7) / 8;                // 782 (8 strips/block)
    const int scatterBlocks = (N_EDGES * 32 + 255) / 256;  // 400000

    // Pre-pack all 12 weight matrices into B-fragment order.
    for (int l = 0; l < N_LAYERS; ++l) {
        const int base = 2 + 13 * l;
        prep_w_kernel<<<16, 256, 0, stream>>>((const float*)d_in[base + 1],
                                              Wfrag + (size_t)(l * 3 + 0) * 4096);
        prep_w_kernel<<<16, 256, 0, stream>>>((const float*)d_in[base + 5],
                                              Wfrag + (size_t)(l * 3 + 1) * 4096);
        prep_w_kernel<<<16, 256, 0, stream>>>((const float*)d_in[base + 9],
                                              Wfrag + (size_t)(l * 3 + 2) * 4096);
    }

    copy_x_kernel<<<elemBlocks, 256, 0, stream>>>(x, out);

    for (int l = 0; l < N_LAYERS; ++l) {
        const int base = 2 + 13 * l;   // eps,W1,b1,g1,be1,W2,b2,g2,be2,W3,b3,go,bo
        const float* eps = (const float*)d_in[base + 0];
        const float* b1  = (const float*)d_in[base + 2];
        const float* g1  = (const float*)d_in[base + 3];
        const float* be1 = (const float*)d_in[base + 4];
        const float* b2  = (const float*)d_in[base + 6];
        const float* g2  = (const float*)d_in[base + 7];
        const float* be2 = (const float*)d_in[base + 8];
        const float* b3  = (const float*)d_in[base + 10];
        const float* go  = (const float*)d_in[base + 11];
        const float* bo  = (const float*)d_in[base + 12];

        // Layer input features: x for l==0, else previous layer's slab in d_out.
        const float* feat = (l == 0) ? x : (out + (size_t)l * 64);
        const int stride  = (l == 0) ? D : OUT_STRIDE;

        // GIN aggregation: H = (1+eps)*feat ; H[dst] += feat[src]
        init_h_kernel<<<elemBlocks, 256, 0, stream>>>(feat, stride, eps, H);
        scatter_kernel<<<scatterBlocks, 256, 0, stream>>>(feat, stride, src, dst, H);
        cvt_bf16_kernel<<<elemBlocks, 256, 0, stream>>>(H, Abf);

        // Linear1 -> BN -> ReLU
        zero_stats_kernel<<<1, 128, 0, stream>>>(stats);
        gemm64_kernel<<<gemmBlocks, 256, 0, stream>>>(Abf, Wfrag + (size_t)(l * 3 + 0) * 4096,
                                                      b1, T, colsum, colsq);
        bnfin_kernel<<<1, 64, 0, stream>>>(colsum, colsq, g1, be1, scale, shift);
        bnapply_bf16_kernel<<<elemBlocks, 256, 0, stream>>>(T, scale, shift, Abf);

        // Linear2 -> BN -> ReLU
        zero_stats_kernel<<<1, 128, 0, stream>>>(stats);
        gemm64_kernel<<<gemmBlocks, 256, 0, stream>>>(Abf, Wfrag + (size_t)(l * 3 + 1) * 4096,
                                                      b2, H, colsum, colsq);
        bnfin_kernel<<<1, 64, 0, stream>>>(colsum, colsq, g2, be2, scale, shift);
        bnapply_bf16_kernel<<<elemBlocks, 256, 0, stream>>>(H, scale, shift, Abf);

        // Linear3 -> outer BN -> ReLU -> output slab
        zero_stats_kernel<<<1, 128, 0, stream>>>(stats);
        gemm64_kernel<<<gemmBlocks, 256, 0, stream>>>(Abf, Wfrag + (size_t)(l * 3 + 2) * 4096,
                                                      b3, T, colsum, colsq);
        bnfin_kernel<<<1, 64, 0, stream>>>(colsum, colsq, go, bo, scale, shift);
        apply_out_kernel<<<elemBlocks, 256, 0, stream>>>(T, scale, shift, out, (l + 1) * 64);
    }
}
